// BitTransformerBlockLegacy_17978733101756
// MI455X (gfx1250) — 16-bit and 8-bit matrix operand order repaired and checked by reading, compiled, not run on hardware
//
#include <hip/hip_runtime.h>
#include <hip/hip_bf16.h>
#include <cstdint>


typedef __attribute__((ext_vector_type(8)))  int       v8i;
typedef __attribute__((ext_vector_type(8)))  float     v8f;
typedef __attribute__((ext_vector_type(16))) _Float16  v16h;

#define EPS_Q   1e-5f
#define RMS_EPS 1e-6f


__global__ void zero_f32_kernel(float* p, int n) {
    int i = blockIdx.x * blockDim.x + threadIdx.x;
    if (i < n) p[i] = 0.0f;
}

__global__ void absum_kernel(const float* __restrict__ w, long long n, float* __restrict__ acc) {
    __shared__ float red[256];
    float s = 0.0f;
    for (long long i = (long long)blockIdx.x * 256 + threadIdx.x; i < n;
         i += (long long)gridDim.x * 256)
        s += fabsf(w[i]);
    red[threadIdx.x] = s;
    __syncthreads();
    for (int o = 128; o > 0; o >>= 1) {
        if (threadIdx.x < o) red[threadIdx.x] += red[threadIdx.x + o];
        __syncthreads();
    }
    if (threadIdx.x == 0) atomicAdd(acc, red[0]);
}

__global__ void wquant_kernel(const float* __restrict__ w, int8_t* __restrict__ wq,
                              const float* __restrict__ acc, float n, long long total) {
    long long i = (long long)blockIdx.x * 256 + threadIdx.x;
    if (i >= total) return;
    float s = 1.0f / fmaxf(acc[0] / n, EPS_Q);
    float q = rintf(w[i] * s);
    q = fminf(fmaxf(q, -1.0f), 1.0f);
    wq[i] = (int8_t)q;
}

__global__ void row_quant_kernel(const float* __restrict__ xin, int8_t* __restrict__ xq,
                                 float* __restrict__ scl, int Kn) {
    __shared__ float red[256];
    int row = blockIdx.x;
    const float* xr = xin + (size_t)row * Kn;
    float amax = 0.0f;
    for (int j = threadIdx.x; j < Kn; j += 256) amax = fmaxf(amax, fabsf(xr[j]));
    red[threadIdx.x] = amax;
    __syncthreads();
    for (int o = 128; o > 0; o >>= 1) {
        if (threadIdx.x < o) red[threadIdx.x] = fmaxf(red[threadIdx.x], red[threadIdx.x + o]);
        __syncthreads();
    }
    float mx = fmaxf(red[0], EPS_Q);
    float s = 127.0f / mx;
    if (threadIdx.x == 0) scl[row] = mx * (1.0f / 127.0f);
    int8_t* qr = xq + (size_t)row * Kn;
    for (int j = threadIdx.x; j < Kn; j += 256) {
        float q = rintf(xr[j] * s);
        q = fminf(fmaxf(q, -128.0f), 127.0f);
        qr[j] = (int8_t)q;
    }
}

__global__ void adaln_quant_kernel(const float* __restrict__ x, const float* __restrict__ emb,
                                   const float* __restrict__ g, int8_t* __restrict__ xq,
                                   float* __restrict__ scl, int Tn, int Dn) {
    __shared__ float red[256];
    __shared__ float sx[1024];
    int row = blockIdx.x;
    int b   = row / Tn;
    const float* xr = x + (size_t)row * Dn;
    const float* er = emb + (size_t)b * 2 * Dn;

    float ssq = 0.0f;
    for (int j = threadIdx.x; j < Dn; j += 256) { float v = xr[j]; ssq += v * v; }
    red[threadIdx.x] = ssq;
    __syncthreads();
    for (int o = 128; o > 0; o >>= 1) {
        if (threadIdx.x < o) red[threadIdx.x] += red[threadIdx.x + o];
        __syncthreads();
    }
    float rms = rsqrtf(red[0] / (float)Dn + RMS_EPS);
    __syncthreads();

    float amax = 0.0f;
    for (int j = threadIdx.x; j < Dn; j += 256) {
        float v = xr[j] * rms * g[j] * (1.0f + er[j]) + er[Dn + j];
        sx[j] = v;
        amax = fmaxf(amax, fabsf(v));
    }
    red[threadIdx.x] = amax;
    __syncthreads();
    for (int o = 128; o > 0; o >>= 1) {
        if (threadIdx.x < o) red[threadIdx.x] = fmaxf(red[threadIdx.x], red[threadIdx.x + o]);
        __syncthreads();
    }
    float mx = fmaxf(red[0], EPS_Q);
    float s = 127.0f / mx;
    if (threadIdx.x == 0) scl[row] = mx * (1.0f / 127.0f);
    int8_t* qr = xq + (size_t)row * Dn;
    for (int j = threadIdx.x; j < Dn; j += 256) {
        float q = rintf(sx[j] * s);
        q = fminf(fmaxf(q, -128.0f), 127.0f);
        qr[j] = (int8_t)q;
    }
}

__global__ void vec_bitlinear_kernel(const int8_t* __restrict__ cq, const float* __restrict__ c_scl,
                                     const int8_t* __restrict__ wq, const float* __restrict__ wacc,
                                     float wn, const float* __restrict__ bias,
                                     float* __restrict__ outp, int Bn, int Nn, int Kn) {
    int idx = blockIdx.x * blockDim.x + threadIdx.x;
    if (idx >= Bn * Nn) return;
    int b = idx / Nn, n = idx % Nn;
    const int8_t* cr = cq + (size_t)b * Kn;
    const int8_t* wr = wq + (size_t)n * Kn;
    int acc = 0;
    for (int k = 0; k < Kn; k += 4) {
        acc += (int)cr[k]     * (int)wr[k]
             + (int)cr[k + 1] * (int)wr[k + 1]
             + (int)cr[k + 2] * (int)wr[k + 2]
             + (int)cr[k + 3] * (int)wr[k + 3];
    }
    float ws = fmaxf(wacc[0] / wn, EPS_Q);
    outp[idx] = (float)acc * c_scl[b] * ws + bias[n];
}

template <int MODE>
__global__ void gemm_i8_kernel(const int8_t* __restrict__ Aq, const float* __restrict__ a_scl,
                               const int8_t* __restrict__ Wq, const float* __restrict__ wacc,
                               float wn, const float* __restrict__ bias,
                               const float* __restrict__ resid, float* __restrict__ outp,
                               int M, int N, int K) {
    int lane  = threadIdx.x & 31;
    int wave  = threadIdx.x >> 5;
    int tm    = blockIdx.x * 2;
    int tn    = (blockIdx.y * 8 + wave) * 4;
    if (tn * 16 >= N) return;
    int half_ = lane >> 4;
    int l15   = lane & 15;

    const int8_t* arow0 = Aq + (size_t)(tm * 16 + l15) * K;
    const int8_t* arow1 = arow0 + (size_t)16 * K;
    const int8_t* wrow0 = Wq + (size_t)(tn * 16 + l15) * K;

    v8i acc[2][4];
#pragma unroll
    for (int mi = 0; mi < 2; ++mi)
#pragma unroll
        for (int ni = 0; ni < 4; ++ni)
#pragma unroll
            for (int r = 0; r < 8; ++r) acc[mi][ni][r] = 0;

    for (int k0 = 0; k0 < K; k0 += 64) {
        v8i a0, a1;
#pragma unroll
        for (int j = 0; j < 8; ++j) {
            int ka = k0 + ((j >> 1) << 4) + (j & 1) * 4 + 8 * half_;
            a0[j] = *(const int*)(arow0 + ka);
            a1[j] = *(const int*)(arow1 + ka);
        }
        if (k0 + 64 < K) {
            __builtin_prefetch(arow0 + k0 + 64, 0, 0);
            __builtin_prefetch(arow1 + k0 + 64, 0, 0);
        }
#pragma unroll
        for (int ni = 0; ni < 4; ++ni) {
            const int8_t* wrow = wrow0 + (size_t)(ni * 16) * K;
            v8i b;
#pragma unroll
            for (int j = 0; j < 8; ++j) {
                int kb = k0 + ((j >> 1) << 4) + (j & 1) * 4 + 8 * half_;
                b[j] = *(const int*)(wrow + kb);
            }
            acc[0][ni] = __builtin_amdgcn_wmma_i32_16x16x64_iu8(true, a0, true, b, acc[0][ni], false, false);
            acc[1][ni] = __builtin_amdgcn_wmma_i32_16x16x64_iu8(true, a1, true, b, acc[1][ni], false, false);
        }
    }

    float ws = fmaxf(wacc[0] / wn, EPS_Q);
#pragma unroll
    for (int mi = 0; mi < 2; ++mi)
#pragma unroll
        for (int ni = 0; ni < 4; ++ni)
#pragma unroll
            for (int r = 0; r < 8; ++r) {
                int row = (tm + mi) * 16 + r + 8 * half_;
                int col = (tn + ni) * 16 + l15;
                float val = (float)acc[mi][ni][r] * a_scl[row] * ws + bias[col];
                if (MODE == 1) val += resid[(size_t)row * N + col];
                if (MODE == 2) val = 0.5f * val * (1.0f + erff(val * 0.70710678118f));
                outp[(size_t)row * N + col] = val;
            }
}


__global__ void qkv_split_kernel(const float* __restrict__ qkv, _Float16* __restrict__ q,
                                 _Float16* __restrict__ k, _Float16* __restrict__ v,
                                 int Bn, int Tn) {
    const int D = 1024, H = 16, HD = 64;
    long long idx = (long long)blockIdx.x * 256 + threadIdx.x;
    long long total = (long long)Bn * Tn * 3 * D;
    if (idx >= total) return;
    int  n   = (int)(idx % (3 * D));
    long long row = idx / (3 * D);
    int  b = (int)(row / Tn), t = (int)(row % Tn);
    int  i = n / D, rem = n % D, h = rem / HD, d = rem % HD;
    _Float16 val = (_Float16)qkv[idx];
    _Float16* dst = (i == 0) ? q : (i == 1) ? k : v;
    dst[(((size_t)b * H + h) * Tn + t) * HD + d] = val;
}


__global__ void __launch_bounds__(256)
flash_attn_kernel(const _Float16* __restrict__ Q, const _Float16* __restrict__ Kh,
                  const _Float16* __restrict__ Vh, float* __restrict__ O, int Tn) {
    const int HD = 64, H = 16, D = 1024;
    int nqb = Tn / 128;
    int qb  = blockIdx.x % nqb;
    int bh  = blockIdx.x / nqb;
    const _Float16* q = Q + (size_t)bh * Tn * HD;
    const _Float16* k = Kh + (size_t)bh * Tn * HD;
    const _Float16* v = Vh + (size_t)bh * Tn * HD;

    int tid   = threadIdx.x;
    int wave  = tid >> 5;
    int lane  = tid & 31;
    int half_ = lane >> 4, l15 = lane & 15;
    int qt    = qb * 8 + wave;

    __shared__ _Float16 sK[32 * 64];
    __shared__ _Float16 sVt[64 * 32];
    __shared__ float    sS[8][16 * 32];
    __shared__ _Float16 sP[8][16 * 32];
    __shared__ float    sAlpha[8][16], sM[8][16], sL[8][16], sInv[8][16];

    if (lane < 16) { sM[wave][lane] = -1e30f; sL[wave][lane] = 0.0f; }

    v16h qa[2];
#pragma unroll
    for (int c = 0; c < 2; ++c)
#pragma unroll
        for (int e = 0; e < 16; ++e) {
            int kk = c * 32 + 16 * half_ + e;
            qa[c][e] = q[(size_t)(qt * 16 + l15) * HD + kk];
        }

    v8f o_acc[4];
#pragma unroll
    for (int c = 0; c < 4; ++c)
#pragma unroll
        for (int r = 0; r < 8; ++r) o_acc[c][r] = 0.0f;

    int krow = tid >> 3;
    int kd0  = (tid & 7) * 8;
    unsigned int sk_base = (unsigned int)(uintptr_t)(&sK[0]);
    __syncthreads();

    for (int kv0 = 0; kv0 < Tn; kv0 += 32) {
        {
            unsigned long long ga =
                (unsigned long long)(uintptr_t)(k + (size_t)(kv0 + krow) * HD + kd0);
            unsigned int ldst = sk_base + (unsigned int)tid * 16u;
            asm volatile("global_load_async_to_lds_b128 %0, %1, off"
                         :: "v"(ldst), "v"(ga) : "memory");
        }
        {
            union { uint4 u; _Float16 h[8]; } pk;
            pk.u = *(const uint4*)(v + (size_t)(kv0 + krow) * HD + kd0);
#pragma unroll
            for (int j = 0; j < 8; ++j) sVt[(kd0 + j) * 32 + krow] = pk.h[j];
        }
        asm volatile("s_wait_asynccnt 0x0" ::: "memory");
        __syncthreads();

#pragma unroll
        for (int nt = 0; nt < 2; ++nt) {
            v8f sacc;
#pragma unroll
            for (int r = 0; r < 8; ++r) sacc[r] = 0.0f;
#pragma unroll
            for (int c = 0; c < 2; ++c) {
                v16h kb;
#pragma unroll
                for (int e = 0; e < 16; ++e)
                    kb[e] = sK[(nt * 16 + l15) * 64 + c * 32 + 16 * half_ + e];
                sacc = __builtin_amdgcn_wmma_f32_16x16x32_f16(false, qa[c], false, kb,
                                                              (short)0, sacc, false, false);
            }
#pragma unroll
            for (int r = 0; r < 8; ++r)
                sS[wave][(r + 8 * half_) * 32 + nt * 16 + l15] = sacc[r] * 0.125f;
        }
        __syncthreads();

        if (lane < 16) {
            int row = lane;
            float m = sM[wave][row], rowmax = m;
            for (int j = 0; j < 32; ++j) rowmax = fmaxf(rowmax, sS[wave][row * 32 + j]);
            float alpha = __expf(m - rowmax);
            float rs = 0.0f;
            for (int j = 0; j < 32; ++j) {
                float p = __expf(sS[wave][row * 32 + j] - rowmax);
                sP[wave][row * 32 + j] = (_Float16)p;
                rs += p;
            }
            sM[wave][row] = rowmax;
            sL[wave][row] = sL[wave][row] * alpha + rs;
            sAlpha[wave][row] = alpha;
        }
        __syncthreads();

#pragma unroll
        for (int c = 0; c < 4; ++c)
#pragma unroll
            for (int r = 0; r < 8; ++r) o_acc[c][r] *= sAlpha[wave][r + 8 * half_];

        v16h pa;
#pragma unroll
        for (int e = 0; e < 16; ++e) {
            int kk = 16 * half_ + e;
            pa[e] = sP[wave][l15 * 32 + kk];
        }

#pragma unroll
        for (int c = 0; c < 4; ++c) {
            v16h vb;
#pragma unroll
            for (int e = 0; e < 16; ++e)
                vb[e] = sVt[(c * 16 + l15) * 32 + 16 * half_ + e];
            o_acc[c] = __builtin_amdgcn_wmma_f32_16x16x32_f16(false, pa, false, vb,
                                                              (short)0, o_acc[c], false, false);
        }
        __syncthreads();
    }

    if (lane < 16) sInv[wave][lane] = 1.0f / sL[wave][lane];
    __syncthreads();

    int b = bh / H, h = bh % H;
#pragma unroll
    for (int c = 0; c < 4; ++c)
#pragma unroll
        for (int r = 0; r < 8; ++r) {
            int row = r + 8 * half_;
            int col = c * 16 + l15;
            O[((size_t)b * Tn + qt * 16 + row) * D + h * HD + col] = o_acc[c][r] * sInv[wave][row];
        }
}


static inline size_t aup(size_t x) { return (x + 255) & ~(size_t)255; }

extern "C" void kernel_launch(void* const* d_in, const int* in_sizes, int n_in,
                              void* d_out, int out_size, void* d_ws, size_t ws_size,
                              hipStream_t stream) {
    (void)in_sizes; (void)n_in; (void)out_size; (void)ws_size;
    constexpr int B = 2, T = 2048, D = 1024, H = 16, FF = 4096, CD = 1024;
    constexpr int M = B * T;

    const float* x      = (const float*)d_in[0];
    const float* c      = (const float*)d_in[1];
    const float* g1     = (const float*)d_in[2];
    const float* w_ada1 = (const float*)d_in[3];
    const float* b_ada1 = (const float*)d_in[4];
    const float* w_qkv  = (const float*)d_in[5];
    const float* b_qkv  = (const float*)d_in[6];
    const float* w_proj = (const float*)d_in[7];
    const float* b_proj = (const float*)d_in[8];
    const float* g2     = (const float*)d_in[9];
    const float* w_ada2 = (const float*)d_in[10];
    const float* b_ada2 = (const float*)d_in[11];
    const float* w_fc1  = (const float*)d_in[12];
    const float* b_fc1  = (const float*)d_in[13];
    const float* w_fc2  = (const float*)d_in[14];
    const float* b_fc2  = (const float*)d_in[15];
    float* out = (float*)d_out;

    char* ws = (char*)d_ws;
    size_t cur = 0;
    auto alloc = [&](size_t bytes) -> char* { char* p = ws + cur; cur = aup(cur + bytes); return p; };

    float*    scal    = (float*)alloc(64 * sizeof(float));
    int8_t*   cq      = (int8_t*)alloc((size_t)B * CD);
    float*    c_scl   = (float*)alloc(B * sizeof(float));
    int8_t*   wq_ada1 = (int8_t*)alloc((size_t)2 * D * CD);
    int8_t*   wq_qkv  = (int8_t*)alloc((size_t)3 * D * D);
    int8_t*   wq_proj = (int8_t*)alloc((size_t)D * D);
    int8_t*   wq_ada2 = (int8_t*)alloc((size_t)2 * D * CD);
    int8_t*   wq_fc1  = (int8_t*)alloc((size_t)FF * D);
    int8_t*   wq_fc2  = (int8_t*)alloc((size_t)D * FF);
    float*    emb1    = (float*)alloc((size_t)B * 2 * D * sizeof(float));
    float*    emb2    = (float*)alloc((size_t)B * 2 * D * sizeof(float));
    int8_t*   xn1q    = (int8_t*)alloc((size_t)M * D);
    float*    xn1_scl = (float*)alloc(M * sizeof(float));
    float*    qkvbuf  = (float*)alloc((size_t)M * 3 * D * sizeof(float));
    _Float16* qf      = (_Float16*)alloc((size_t)M * D * sizeof(_Float16));
    _Float16* kf      = (_Float16*)alloc((size_t)M * D * sizeof(_Float16));
    _Float16* vf      = (_Float16*)alloc((size_t)M * D * sizeof(_Float16));
    float*    obuf    = (float*)alloc((size_t)M * D * sizeof(float));
    int8_t*   oq      = (int8_t*)alloc((size_t)M * D);
    float*    o_scl   = (float*)alloc(M * sizeof(float));
    float*    x1      = (float*)alloc((size_t)M * D * sizeof(float));
    int8_t*   xn2q    = (int8_t*)alloc((size_t)M * D);
    float*    xn2_scl = (float*)alloc(M * sizeof(float));
    float*    hbuf    = (float*)alloc((size_t)M * FF * sizeof(float));
    int8_t*   hq      = (int8_t*)alloc((size_t)M * FF);
    float*    h_scl   = (float*)alloc(M * sizeof(float));

    const long long n_ada  = (long long)2 * D * CD;
    const long long n_qkv  = (long long)3 * D * D;
    const long long n_proj = (long long)D * D;
    const long long n_fc   = (long long)FF * D;

    zero_f32_kernel<<<1, 64, 0, stream>>>(scal, 64);

    absum_kernel<<<1024, 256, 0, stream>>>(w_ada1, n_ada,  &scal[0]);
    absum_kernel<<<1024, 256, 0, stream>>>(w_qkv,  n_qkv,  &scal[1]);
    absum_kernel<<<1024, 256, 0, stream>>>(w_proj, n_proj, &scal[2]);
    absum_kernel<<<1024, 256, 0, stream>>>(w_ada2, n_ada,  &scal[3]);
    absum_kernel<<<1024, 256, 0, stream>>>(w_fc1,  n_fc,   &scal[4]);
    absum_kernel<<<1024, 256, 0, stream>>>(w_fc2,  n_fc,   &scal[5]);
    wquant_kernel<<<(int)((n_ada  + 255) / 256), 256, 0, stream>>>(w_ada1, wq_ada1, &scal[0], (float)n_ada,  n_ada);
    wquant_kernel<<<(int)((n_qkv  + 255) / 256), 256, 0, stream>>>(w_qkv,  wq_qkv,  &scal[1], (float)n_qkv,  n_qkv);
    wquant_kernel<<<(int)((n_proj + 255) / 256), 256, 0, stream>>>(w_proj, wq_proj, &scal[2], (float)n_proj, n_proj);
    wquant_kernel<<<(int)((n_ada  + 255) / 256), 256, 0, stream>>>(w_ada2, wq_ada2, &scal[3], (float)n_ada,  n_ada);
    wquant_kernel<<<(int)((n_fc   + 255) / 256), 256, 0, stream>>>(w_fc1,  wq_fc1,  &scal[4], (float)n_fc,   n_fc);
    wquant_kernel<<<(int)((n_fc   + 255) / 256), 256, 0, stream>>>(w_fc2,  wq_fc2,  &scal[5], (float)n_fc,   n_fc);

    row_quant_kernel<<<B, 256, 0, stream>>>(c, cq, c_scl, CD);
    vec_bitlinear_kernel<<<(B * 2 * D + 255) / 256, 256, 0, stream>>>(
        cq, c_scl, wq_ada1, &scal[0], (float)n_ada, b_ada1, emb1, B, 2 * D, CD);
    vec_bitlinear_kernel<<<(B * 2 * D + 255) / 256, 256, 0, stream>>>(
        cq, c_scl, wq_ada2, &scal[3], (float)n_ada, b_ada2, emb2, B, 2 * D, CD);

    adaln_quant_kernel<<<M, 256, 0, stream>>>(x, emb1, g1, xn1q, xn1_scl, T, D);
    gemm_i8_kernel<0><<<dim3(M / 32, (3 * D) / 512), 256, 0, stream>>>(
        xn1q, xn1_scl, wq_qkv, &scal[1], (float)n_qkv, b_qkv, nullptr, qkvbuf, M, 3 * D, D);
    qkv_split_kernel<<<(int)(((long long)M * 3 * D + 255) / 256), 256, 0, stream>>>(
        qkvbuf, qf, kf, vf, B, T);

    flash_attn_kernel<<<B * H * (T / 128), 256, 0, stream>>>(qf, kf, vf, obuf, T);

    row_quant_kernel<<<M, 256, 0, stream>>>(obuf, oq, o_scl, D);
    gemm_i8_kernel<1><<<dim3(M / 32, D / 512), 256, 0, stream>>>(
        oq, o_scl, wq_proj, &scal[2], (float)n_proj, b_proj, x, x1, M, D, D);

    adaln_quant_kernel<<<M, 256, 0, stream>>>(x1, emb2, g2, xn2q, xn2_scl, T, D);
    gemm_i8_kernel<2><<<dim3(M / 32, FF / 512), 256, 0, stream>>>(
        xn2q, xn2_scl, wq_fc1, &scal[4], (float)n_fc, b_fc1, nullptr, hbuf, M, FF, D);
    row_quant_kernel<<<M, 256, 0, stream>>>(hbuf, hq, h_scl, FF);
    gemm_i8_kernel<1><<<dim3(M / 32, D / 512), 256, 0, stream>>>(
        hq, h_scl, wq_fc2, &scal[5], (float)n_fc, b_fc2, x1, out, M, D, FF);
}
